// TransformerMoEDecoderLayer_49108656063223
// MI455X (gfx1250) — matrix operand order repaired and checked by reading, not compiled, not run on hardware
//
#include <hip/hip_runtime.h>
#include <hip/hip_bf16.h>

// ---------------------------------------------------------------------------
// TransformerMoEDecoderLayer for MI455X (gfx1250, wave32, WMMA).
// All GEMMs use v_wmma_f32_16x16x32_bf16 (fp32 inputs converted to bf16 with
// native v_cvt_pk_bf16_f32, f32 accumulate). Attention is flash-style with
// DPP16 cross-lane softmax reductions (no ds_bpermute round trips).
// MoE routing is a deterministic ordered scan (matches jnp.cumsum semantics).
// ---------------------------------------------------------------------------

typedef __attribute__((ext_vector_type(16))) __bf16 bf16x16;
typedef __attribute__((ext_vector_type(2)))  __bf16 bf16x2;
typedef __attribute__((ext_vector_type(8)))  float  f32x8;

union BfFrag { unsigned int u[8]; bf16x16 v; };

// Native conversion: clang lowers (__bf16) casts to v_cvt_pk_bf16_f32 pairs.
__device__ __forceinline__ unsigned int pack2bf(float a, float b) {
  union { bf16x2 h; unsigned int u; } p;
  p.h[0] = (__bf16)a;
  p.h[1] = (__bf16)b;
  return p.u;
}

// ---------------------------------------------------------------------------
// 16-lane (half-row) butterfly reductions via DPP16 — pure VALU, co-executes
// with WMMA, no DS pipe usage (vs __shfl_xor -> ds_bpermute_b32).
//   0xB1  = quad_perm(1,0,3,2)  : xor 1
//   0x4E  = quad_perm(2,3,0,1)  : xor 2
//   0x141 = row_half_mirror     : combines quads within 8-lane group
//   0x140 = row_mirror          : combines 8-lane halves within 16-lane row
// ---------------------------------------------------------------------------
template<int CTRL>
__device__ __forceinline__ float dpp_f(float x) {
  return __int_as_float(__builtin_amdgcn_mov_dpp(
      __float_as_int(x), CTRL, 0xF, 0xF, true));
}
__device__ __forceinline__ float row16_max(float x) {
  x = fmaxf(x, dpp_f<0xB1>(x));
  x = fmaxf(x, dpp_f<0x4E>(x));
  x = fmaxf(x, dpp_f<0x141>(x));
  x = fmaxf(x, dpp_f<0x140>(x));
  return x;
}
__device__ __forceinline__ float row16_sum(float x) {
  x += dpp_f<0xB1>(x);
  x += dpp_f<0x4E>(x);
  x += dpp_f<0x141>(x);
  x += dpp_f<0x140>(x);
  return x;
}

// ---------------------------------------------------------------------------
// Tiled WMMA GEMM.  C[M,N] = A[M,K] * op(B) + bias, optional ReLU, batched.
//   BT=true : B is (N,K) row-major  (C = A * B^T)   -- used for x @ W.T
//   BT=false: B is (K,N) row-major  (C = A * B)     -- used for wg / w1 / w2
// Block tile 128x128, K-step 32. 8 waves, each wave computes 32x64
// (2x4 WMMA 16x16 accumulators) -> 8 v_wmma per K-step per wave.
// Fragment layouts per CDNA5 ISA 7.12.2:
//   A 16x32 bf16: lane L: m=L&15, vgpr v holds K = (v&4)*4 + (L>>4)*8 + (v&3)*2 (+1)
//   B 32x16 bf16: lane L: n=L&15, vgpr v holds K = (v&4)*4 + (L>>4)*8 + (v&3)*2 (+1)
//   C 16x16 f32 : lane L: n=L&15, vgpr v holds m = v + (L>>4)*8
// ---------------------------------------------------------------------------
template<bool BT>
__global__ void __launch_bounds__(256) gemm_wmma(
    const float* __restrict__ A, const float* __restrict__ Bm,
    const float* __restrict__ bias, float* __restrict__ C,
    int M, int N, int K, int lda, int ldb, int ldc,
    long long sA, long long sB, long long sBias, long long sC, int relu)
{
  const int z = blockIdx.z;
  A  += (long long)z * sA;
  Bm += (long long)z * sB;
  C  += (long long)z * sC;
  const float* biasz = bias ? (bias + (long long)z * sBias) : nullptr;

  __shared__ __bf16 aS[128 * 32];   // [m][k]
  __shared__ __bf16 bS[128 * 32];   // [n][k]

  const int tid  = threadIdx.x;
  const int lane = tid & 31;
  const int l15  = lane & 15;
  const int lh   = lane >> 4;
  const int wid  = tid >> 5;
  const int wm   = (wid & 3) << 5;          // wave m offset: 0,32,64,96
  const int wn   = (wid >> 2) << 6;         // wave n offset: 0,64

  const int m0 = blockIdx.y * 128;
  const int n0 = blockIdx.x * 128;

  const f32x8 zf = {0.f,0.f,0.f,0.f,0.f,0.f,0.f,0.f};
  f32x8 acc[2][4];
  #pragma unroll
  for (int i = 0; i < 2; ++i)
    #pragma unroll
    for (int j = 0; j < 4; ++j) acc[i][j] = zf;

  for (int kk = 0; kk < K; kk += 32) {
    // ---- stage A tile (128x32 f32 -> bf16 LDS), float4 loads -------------
    #pragma unroll
    for (int i = 0; i < 4; ++i) {
      int s = tid + (i << 8);
      int r = s >> 3;                 // m within tile
      int c = (s & 7) << 2;           // k within tile
      float4 f = make_float4(0.f, 0.f, 0.f, 0.f);
      int gm = m0 + r;
      if (gm < M) {
        const float* ap = A + (long long)gm * lda + kk + c;
        if (kk + 32 < K) __builtin_prefetch(ap + 32, 0, 3);  // global_prefetch_b8
        f = *(const float4*)ap;
      }
      aS[r*32 + c + 0] = (__bf16)f.x;
      aS[r*32 + c + 1] = (__bf16)f.y;
      aS[r*32 + c + 2] = (__bf16)f.z;
      aS[r*32 + c + 3] = (__bf16)f.w;
    }
    // ---- stage B tile -> bS[n][k] ----------------------------------------
    if (BT) {
      #pragma unroll
      for (int i = 0; i < 4; ++i) {
        int s = tid + (i << 8);
        int r = s >> 3;               // n within tile
        int c = (s & 7) << 2;         // k within tile
        float4 f = make_float4(0.f, 0.f, 0.f, 0.f);
        int gn = n0 + r;
        if (gn < N) f = *(const float4*)(Bm + (long long)gn * ldb + kk + c);
        bS[r*32 + c + 0] = (__bf16)f.x;
        bS[r*32 + c + 1] = (__bf16)f.y;
        bS[r*32 + c + 2] = (__bf16)f.z;
        bS[r*32 + c + 3] = (__bf16)f.w;
      }
    } else {
      #pragma unroll
      for (int i = 0; i < 4; ++i) {
        int s = tid + (i << 8);
        int r = s >> 5;               // k within tile (0..31)
        int c = (s & 31) << 2;        // n within tile
        int gk = kk + r;
        int gn = n0 + c;
        float4 f;
        if (gn + 3 < N) {
          f = *(const float4*)(Bm + (long long)gk * ldb + gn);
        } else {
          f.x = (gn + 0 < N) ? Bm[(long long)gk * ldb + gn + 0] : 0.f;
          f.y = (gn + 1 < N) ? Bm[(long long)gk * ldb + gn + 1] : 0.f;
          f.z = (gn + 2 < N) ? Bm[(long long)gk * ldb + gn + 2] : 0.f;
          f.w = (gn + 3 < N) ? Bm[(long long)gk * ldb + gn + 3] : 0.f;
        }
        bS[(c + 0)*32 + r] = (__bf16)f.x;   // transpose to [n][k]
        bS[(c + 1)*32 + r] = (__bf16)f.y;
        bS[(c + 2)*32 + r] = (__bf16)f.z;
        bS[(c + 3)*32 + r] = (__bf16)f.w;
      }
    }
    __syncthreads();

    // ---- fragments + WMMA ------------------------------------------------
    BfFrag a[2], b[4];
    #pragma unroll
    for (int i = 0; i < 2; ++i)
      #pragma unroll
      for (int v = 0; v < 8; ++v) {
        int m  = wm + (i << 4) + l15;
        int k0 = ((v & 4) << 2) + (lh << 3) + ((v & 3) << 1);
        a[i].u[v] = *(const unsigned int*)&aS[m*32 + k0];
      }
    #pragma unroll
    for (int j = 0; j < 4; ++j)
      #pragma unroll
      for (int v = 0; v < 8; ++v) {
        int n  = wn + (j << 4) + l15;
        int k0 = ((v & 4) << 2) + (lh << 3) + ((v & 3) << 1);
        b[j].u[v] = *(const unsigned int*)&bS[n*32 + k0];
      }
    #pragma unroll
    for (int i = 0; i < 2; ++i)
      #pragma unroll
      for (int j = 0; j < 4; ++j)
        acc[i][j] = __builtin_amdgcn_wmma_f32_16x16x32_bf16(
            false, a[i].v, false, b[j].v, (short)0, acc[i][j], false, false);
    __syncthreads();
  }

  // ---- epilogue: bias (+ReLU) + store ------------------------------------
  #pragma unroll
  for (int j = 0; j < 4; ++j) {
    int gn = n0 + wn + (j << 4) + l15;
    float bv = (biasz && gn < N) ? biasz[gn] : 0.f;
    #pragma unroll
    for (int i = 0; i < 2; ++i)
      #pragma unroll
      for (int v = 0; v < 8; ++v) {
        int gm = m0 + wm + (i << 4) + v + (lh << 3);
        if (gm < M && gn < N) {
          float val = acc[i][j][v] + bv;
          if (relu) val = fmaxf(val, 0.f);
          C[(long long)gm * ldc + gn] = val;
        }
      }
  }
}

// ---------------------------------------------------------------------------
// Flash attention, one wave per (16 q-rows, head, batch).  hd = 64.
// Token layout: row t = s*Bsz + b, feature col = h*64 + k, row stride ld*.
// Per 32-key chunk: 4 WMMA for Q*K^T (2 key groups x 2 hd halves) +
// DPP-based online softmax + 4 WMMA for P*V (P round-trips LDS to convert the
// C-layout fragment into an A-layout bf16 fragment).
// ---------------------------------------------------------------------------
__global__ void __launch_bounds__(32) attn_fwd(
    const float* __restrict__ Q, int ldq,
    const float* __restrict__ Kp, int ldk,
    const float* __restrict__ Vp, int ldv,
    float* __restrict__ O, int ldo,
    int Sk, int Bsz, float scale)
{
  const int lane = threadIdx.x;
  const int l15  = lane & 15;
  const int lh   = lane >> 4;
  const int h    = blockIdx.y;
  const int b    = blockIdx.z;
  const int s0   = blockIdx.x << 4;

  __shared__ __bf16 pbuf[16 * 32];   // P tile, bf16 [m][k]

  // Q fragments (16 x 64 -> two 16x32 A fragments)
  BfFrag aq[2];
  {
    long long qoff = ((long long)(s0 + l15) * Bsz + b) * ldq + h * 64;
    #pragma unroll
    for (int half = 0; half < 2; ++half)
      #pragma unroll
      for (int v = 0; v < 8; ++v) {
        int k0 = ((v & 4) << 2) + (lh << 3) + ((v & 3) << 1) + (half << 5);
        float2 f = *(const float2*)(Q + qoff + k0);
        aq[half].u[v] = pack2bf(f.x, f.y);
      }
  }

  const f32x8 zf = {0.f,0.f,0.f,0.f,0.f,0.f,0.f,0.f};
  f32x8 acc[4] = {zf, zf, zf, zf};
  float rmax[8], rsum[8];
  #pragma unroll
  for (int v = 0; v < 8; ++v) { rmax[v] = -INFINITY; rsum[v] = 0.f; }

  for (int kc = 0; kc < Sk; kc += 32) {
    // ---- scores: two 16-key groups --------------------------------------
    f32x8 sc[2];
    #pragma unroll
    for (int g = 0; g < 2; ++g) {
      BfFrag bk0, bk1;
      long long koff = ((long long)(kc + (g << 4) + l15) * Bsz + b) * ldk + h * 64;
      #pragma unroll
      for (int v = 0; v < 8; ++v) {
        int k0 = ((v & 4) << 2) + (lh << 3) + ((v & 3) << 1);
        float2 f0 = *(const float2*)(Kp + koff + k0);
        float2 f1 = *(const float2*)(Kp + koff + 32 + k0);
        bk0.u[v] = pack2bf(f0.x, f0.y);
        bk1.u[v] = pack2bf(f1.x, f1.y);
      }
      f32x8 c = zf;
      c = __builtin_amdgcn_wmma_f32_16x16x32_bf16(false, aq[0].v, false, bk0.v, (short)0, c, false, false);
      c = __builtin_amdgcn_wmma_f32_16x16x32_bf16(false, aq[1].v, false, bk1.v, (short)0, c, false, false);
      sc[g] = c;
    }
    // ---- online softmax (row = v + 8*lh); DPP16 reductions ---------------
    float p0v[8], p1v[8];
    #pragma unroll
    for (int v = 0; v < 8; ++v) {
      float a0 = sc[0][v] * scale;
      float a1 = sc[1][v] * scale;
      float m  = row16_max(fmaxf(a0, a1));
      float nm = fmaxf(rmax[v], m);
      float al = __expf(rmax[v] - nm);
      float p0 = __expf(a0 - nm);
      float p1 = __expf(a1 - nm);
      float ss = row16_sum(p0 + p1);
      rsum[v] = rsum[v] * al + ss;
      rmax[v] = nm;
      p0v[v] = p0; p1v[v] = p1;
      #pragma unroll
      for (int j = 0; j < 4; ++j) acc[j][v] *= al;
    }
    // ---- P -> LDS (C-layout) -> A-layout bf16 fragment -------------------
    #pragma unroll
    for (int v = 0; v < 8; ++v) {
      int m = (lh << 3) + v;
      pbuf[m*32 + l15]      = (__bf16)p0v[v];
      pbuf[m*32 + 16 + l15] = (__bf16)p1v[v];
    }
    __syncthreads();
    BfFrag pa;
    #pragma unroll
    for (int v = 0; v < 8; ++v) {
      int k0 = ((v & 4) << 2) + (lh << 3) + ((v & 3) << 1);
      pa.u[v] = *(const unsigned int*)&pbuf[l15*32 + k0];
    }
    // ---- P * V -----------------------------------------------------------
    #pragma unroll
    for (int j = 0; j < 4; ++j) {
      BfFrag bv;
      #pragma unroll
      for (int v = 0; v < 8; ++v) {
        int kx = ((v & 4) << 2) + (lh << 3) + ((v & 3) << 1);
        long long r0 = ((long long)(kc + kx)     * Bsz + b) * ldv + h * 64 + (j << 4) + l15;
        long long r1 = ((long long)(kc + kx + 1) * Bsz + b) * ldv + h * 64 + (j << 4) + l15;
        bv.u[v] = pack2bf(Vp[r0], Vp[r1]);
      }
      acc[j] = __builtin_amdgcn_wmma_f32_16x16x32_bf16(false, pa.v, false, bv.v, (short)0, acc[j], false, false);
    }
    __syncthreads();
  }

  // ---- normalize + store --------------------------------------------------
  #pragma unroll
  for (int v = 0; v < 8; ++v) {
    float inv = (rsum[v] > 0.f) ? 1.f / rsum[v] : 0.f;
    #pragma unroll
    for (int j = 0; j < 4; ++j) {
      long long ooff = ((long long)(s0 + (lh << 3) + v) * Bsz + b) * ldo + h * 64 + (j << 4) + l15;
      O[ooff] = acc[j][v] * inv;
    }
  }
}

// ---------------------------------------------------------------------------
// y[t,:] = LayerNorm(x[t,:] + r[t,:]) * g + b     (D == 1024, block == 256)
// ---------------------------------------------------------------------------
__global__ void __launch_bounds__(256) add_ln_kernel(
    const float* __restrict__ X, const float* __restrict__ R,
    const float* __restrict__ G, const float* __restrict__ Bt,
    float* __restrict__ Y, int D)
{
  const int t = blockIdx.x;
  const int tid = threadIdx.x;
  const float* x = X + (long long)t * D;
  const float* r = R + (long long)t * D;
  float* y = Y + (long long)t * D;
  float vals[4];
  float s = 0.f, s2 = 0.f;
  #pragma unroll
  for (int i = 0; i < 4; ++i) {
    int idx = tid + (i << 8);
    float v = x[idx] + r[idx];
    vals[i] = v; s += v; s2 += v * v;
  }
  #pragma unroll
  for (int o = 1; o < 32; o <<= 1) { s += __shfl_xor(s, o, 32); s2 += __shfl_xor(s2, o, 32); }
  __shared__ float ws1[8], ws2[8];
  if ((tid & 31) == 0) { ws1[tid >> 5] = s; ws2[tid >> 5] = s2; }
  __syncthreads();
  s = 0.f; s2 = 0.f;
  #pragma unroll
  for (int w = 0; w < 8; ++w) { s += ws1[w]; s2 += ws2[w]; }
  float mean = s / (float)D;
  float var  = s2 / (float)D - mean * mean;
  float rstd = rsqrtf(var + 1e-5f);
  #pragma unroll
  for (int i = 0; i < 4; ++i) {
    int idx = tid + (i << 8);
    y[idx] = (vals[i] - mean) * rstd * G[idx] + Bt[idx];
  }
}

// ---------------------------------------------------------------------------
// MoE routing (deterministic, matches reference cumsum ordering).
// ---------------------------------------------------------------------------
__global__ void route_prep(const float* __restrict__ logits,
                           int* __restrict__ idx1, int* __restrict__ idx2,
                           float* __restrict__ ga1, float* __restrict__ ga2, int T)
{
  int t = blockIdx.x * blockDim.x + threadIdx.x;
  if (t >= T) return;
  const float* l = logits + (long long)t * 16;
  float L[16], ex[16];
  float mx = -INFINITY;
  #pragma unroll
  for (int i = 0; i < 16; ++i) { L[i] = l[i]; mx = fmaxf(mx, L[i]); }
  float se = 0.f;
  #pragma unroll
  for (int i = 0; i < 16; ++i) { ex[i] = __expf(L[i] - mx); se += ex[i]; }
  int i1 = 0; float b1v = L[0];
  #pragma unroll
  for (int i = 1; i < 16; ++i) if (L[i] > b1v) { b1v = L[i]; i1 = i; }
  int i2 = -1; float b2v = -INFINITY;
  #pragma unroll
  for (int i = 0; i < 16; ++i) if (i != i1 && L[i] > b2v) { b2v = L[i]; i2 = i; }
  idx1[t] = i1; idx2[t] = i2;
  ga1[t] = ex[i1] / se; ga2[t] = ex[i2] / se;
}

__global__ void __launch_bounds__(256) route_scan(
    const int* __restrict__ idx1, const int* __restrict__ idx2,
    int* __restrict__ loc1, int* __restrict__ pre2, int* __restrict__ total1, int T)
{
  __shared__ int se1[256], se2[256];
  __shared__ int cnt1[16], cnt2[16];
  int tid = threadIdx.x;
  if (tid < 16) { cnt1[tid] = 0; cnt2[tid] = 0; }
  __syncthreads();
  for (int base = 0; base < T; base += 256) {
    int t = base + tid;
    int e1 = idx1[t], e2 = idx2[t];
    se1[tid] = e1; se2[tid] = e2;
    __syncthreads();
    int p1 = 0, p2 = 0;
    for (int j = 0; j < tid; ++j) { p1 += (se1[j] == e1); p2 += (se2[j] == e2); }
    loc1[t] = cnt1[e1] + p1;
    pre2[t] = cnt2[e2] + p2;
    __syncthreads();
    if (tid < 16) {
      int c1 = 0, c2 = 0;
      for (int j = 0; j < 256; ++j) { c1 += (se1[j] == tid); c2 += (se2[j] == tid); }
      cnt1[tid] += c1; cnt2[tid] += c2;
    }
    __syncthreads();
  }
  if (tid < 16) total1[tid] = cnt1[tid];
}

__global__ void route_final(const int* __restrict__ idx1, const int* __restrict__ idx2,
                            const int* __restrict__ loc1, const int* __restrict__ pre2,
                            const int* __restrict__ total1,
                            const float* __restrict__ ga1, const float* __restrict__ ga2,
                            int* __restrict__ slot1, int* __restrict__ slot2,
                            float* __restrict__ g1o, float* __restrict__ g2o, int T, int Cap)
{
  int t = blockIdx.x * blockDim.x + threadIdx.x;
  if (t >= T) return;
  int e1 = idx1[t], e2 = idx2[t];
  int l1 = loc1[t];
  int l2 = pre2[t] + total1[e2];
  bool k1 = l1 < Cap, k2 = l2 < Cap;
  float g1 = k1 ? ga1[t] : 0.f;
  float g2 = k2 ? ga2[t] : 0.f;
  float den = fmaxf(g1 + g2, 1.1920929e-07f);
  g1 /= den; g2 /= den;
  slot1[t] = k1 ? (e1 * Cap + l1) : -1;
  slot2[t] = k2 ? (e2 * Cap + l2) : -1;
  g1o[t] = g1; g2o[t] = g2;
}

__global__ void zero_f4(float4* __restrict__ p, int n4) {
  int i = blockIdx.x * blockDim.x + threadIdx.x;
  if (i < n4) p[i] = make_float4(0.f, 0.f, 0.f, 0.f);
}

// grid = T, block = 256 (D == 1024 -> one float4 per thread)
__global__ void __launch_bounds__(256) moe_gather(
    const float* __restrict__ X, const int* __restrict__ s1p, const int* __restrict__ s2p,
    float* __restrict__ xe, int D)
{
  int t = blockIdx.x, tid = threadIdx.x;
  int s1 = s1p[t], s2 = s2p[t];
  float4 v = ((const float4*)(X + (long long)t * D))[tid];
  if (s1 >= 0) ((float4*)(xe + (long long)s1 * D))[tid] = v;
  if (s2 >= 0) ((float4*)(xe + (long long)s2 * D))[tid] = v;
}

__global__ void __launch_bounds__(256) moe_combine(
    const float* __restrict__ ye, const int* __restrict__ s1p, const int* __restrict__ s2p,
    const float* __restrict__ g1p, const float* __restrict__ g2p,
    float* __restrict__ out, int D)
{
  int t = blockIdx.x, tid = threadIdx.x;
  int s1 = s1p[t], s2 = s2p[t];
  float g1 = g1p[t], g2 = g2p[t];
  float4 a = make_float4(0.f, 0.f, 0.f, 0.f);
  if (s1 >= 0) {
    float4 v = ((const float4*)(ye + (long long)s1 * D))[tid];
    a.x += g1 * v.x; a.y += g1 * v.y; a.z += g1 * v.z; a.w += g1 * v.w;
  }
  if (s2 >= 0) {
    float4 v = ((const float4*)(ye + (long long)s2 * D))[tid];
    a.x += g2 * v.x; a.y += g2 * v.y; a.z += g2 * v.z; a.w += g2 * v.w;
  }
  ((float4*)(out + (long long)t * D))[tid] = a;
}

// ---------------------------------------------------------------------------
// Orchestration
// ---------------------------------------------------------------------------
extern "C" void kernel_launch(void* const* d_in, const int* in_sizes, int n_in,
                              void* d_out, int out_size, void* d_ws, size_t ws_size,
                              hipStream_t stream)
{
  (void)in_sizes; (void)n_in; (void)out_size; (void)ws_size;
  const float* tgt      = (const float*)d_in[0];
  const float* mem      = (const float*)d_in[1];
  const float* sa_in_w  = (const float*)d_in[2];
  const float* sa_in_b  = (const float*)d_in[3];
  const float* sa_out_w = (const float*)d_in[4];
  const float* sa_out_b = (const float*)d_in[5];
  const float* ca_in_w  = (const float*)d_in[6];
  const float* ca_in_b  = (const float*)d_in[7];
  const float* ca_out_w = (const float*)d_in[8];
  const float* ca_out_b = (const float*)d_in[9];
  const float* wg       = (const float*)d_in[10];
  const float* w1       = (const float*)d_in[11];
  const float* b1       = (const float*)d_in[12];
  const float* w2       = (const float*)d_in[13];
  const float* b2       = (const float*)d_in[14];
  const float* ln1_g    = (const float*)d_in[15];
  const float* ln1_b    = (const float*)d_in[16];
  const float* ln2_g    = (const float*)d_in[17];
  const float* ln2_b    = (const float*)d_in[18];
  const float* ln3_g    = (const float*)d_in[19];
  const float* ln3_b    = (const float*)d_in[20];

  constexpr int S = 1024, Bq = 4, D = 1024, H = 16, F = 4096, E = 16;
  constexpr int T   = S * Bq;            // 4096 target tokens
  constexpr int Tm  = 1024 * Bq;         // 4096 memory tokens
  constexpr int Cap = 2 * T / E;         // 512 capacity per expert
  const float scale = 0.125f;            // 1/sqrt(64)

  // workspace layout (~305 MB of f32)
  float* ws = (float*)d_ws;
  size_t o = 0;
  float* qkv = ws + o; o += (size_t)T * 3 * D;    // also reused for cross q / kv
  float* ctx = ws + o; o += (size_t)T * D;
  float* t2  = ws + o; o += (size_t)T * D;
  float* x1  = ws + o; o += (size_t)T * D;
  float* x2  = ws + o; o += (size_t)T * D;
  float* lgt = ws + o; o += (size_t)T * E;
  float* ga1 = ws + o; o += T;
  float* ga2 = ws + o; o += T;
  float* g1o = ws + o; o += T;
  float* g2o = ws + o; o += T;
  int* idx1   = (int*)(ws + o); o += T;
  int* idx2   = (int*)(ws + o); o += T;
  int* loc1   = (int*)(ws + o); o += T;
  int* pre2   = (int*)(ws + o); o += T;
  int* slot1  = (int*)(ws + o); o += T;
  int* slot2  = (int*)(ws + o); o += T;
  int* total1 = (int*)(ws + o); o += 16;
  float* xe = ws + o; o += (size_t)E * Cap * D;
  float* hb = ws + o; o += (size_t)E * Cap * F;
  float* ye = ws + o; o += (size_t)E * Cap * D;

  dim3 blk256(256), blk32(32);

  // 1) self-attn QKV projection: qkv = tgt @ sa_in_w.T + sa_in_b
  gemm_wmma<true><<<dim3(3 * D / 128, T / 128, 1), blk256, 0, stream>>>(
      tgt, sa_in_w, sa_in_b, qkv, T, 3 * D, D, D, D, 3 * D, 0, 0, 0, 0, 0);
  // 2) self attention -> ctx
  attn_fwd<<<dim3(S / 16, H, Bq), blk32, 0, stream>>>(
      qkv, 3 * D, qkv + D, 3 * D, qkv + 2 * D, 3 * D, ctx, D, S, Bq, scale);
  // 3) out projection
  gemm_wmma<true><<<dim3(D / 128, T / 128, 1), blk256, 0, stream>>>(
      ctx, sa_out_w, sa_out_b, t2, T, D, D, D, D, D, 0, 0, 0, 0, 0);
  // 4) x1 = LN(tgt + t2)
  add_ln_kernel<<<dim3(T), blk256, 0, stream>>>(tgt, t2, ln1_g, ln1_b, x1, D);

  // 5) cross-attn q projection (rows 0..D of ca_in_w)
  gemm_wmma<true><<<dim3(D / 128, T / 128, 1), blk256, 0, stream>>>(
      x1, ca_in_w, ca_in_b, qkv, T, D, D, D, D, D, 0, 0, 0, 0, 0);
  // 6) cross-attn kv projection of memory (rows D..3D of ca_in_w)
  float* kvc = qkv + (size_t)T * D;
  gemm_wmma<true><<<dim3(2 * D / 128, Tm / 128, 1), blk256, 0, stream>>>(
      mem, ca_in_w + (size_t)D * D, ca_in_b + D, kvc, Tm, 2 * D, D, D, D, 2 * D,
      0, 0, 0, 0, 0);
  // 7) cross attention -> ctx
  attn_fwd<<<dim3(S / 16, H, Bq), blk32, 0, stream>>>(
      qkv, D, kvc, 2 * D, kvc + D, 2 * D, ctx, D, 1024, Bq, scale);
  // 8) out projection
  gemm_wmma<true><<<dim3(D / 128, T / 128, 1), blk256, 0, stream>>>(
      ctx, ca_out_w, ca_out_b, t2, T, D, D, D, D, D, 0, 0, 0, 0, 0);
  // 9) x2 = LN(x1 + t2)
  add_ln_kernel<<<dim3(T), blk256, 0, stream>>>(x1, t2, ln2_g, ln2_b, x2, D);

  // 10) gating logits = x2 @ wg   (N = 16, guarded tile)
  gemm_wmma<false><<<dim3(1, T / 128, 1), blk256, 0, stream>>>(
      x2, wg, nullptr, lgt, T, E, D, D, E, E, 0, 0, 0, 0, 0);
  // 11-13) deterministic top-2 routing with capacity
  route_prep<<<dim3(T / 256), blk256, 0, stream>>>(lgt, idx1, idx2, ga1, ga2, T);
  route_scan<<<dim3(1), blk256, 0, stream>>>(idx1, idx2, loc1, pre2, total1, T);
  route_final<<<dim3(T / 256), blk256, 0, stream>>>(
      idx1, idx2, loc1, pre2, total1, ga1, ga2, slot1, slot2, g1o, g2o, T, Cap);
  // 14) zero expert input buffer, 15) dispatch tokens
  {
    int n4 = E * Cap * D / 4;
    zero_f4<<<dim3((n4 + 255) / 256), blk256, 0, stream>>>((float4*)xe, n4);
  }
  moe_gather<<<dim3(T), blk256, 0, stream>>>(x2, slot1, slot2, xe, D);
  // 16) expert FFN layer 1 (batched over E, ReLU)
  gemm_wmma<false><<<dim3(F / 128, Cap / 128, E), blk256, 0, stream>>>(
      xe, w1, b1, hb, Cap, F, D, D, F, F,
      (long long)Cap * D, (long long)D * F, (long long)F, (long long)Cap * F, 1);
  // 17) expert FFN layer 2 (batched over E)
  gemm_wmma<false><<<dim3(D / 128, Cap / 128, E), blk256, 0, stream>>>(
      hb, w2, b2, ye, Cap, D, F, F, D, D,
      (long long)Cap * F, (long long)F * D, (long long)D, (long long)Cap * D, 0);
  // 18) weighted combine, 19) final LN -> d_out
  moe_combine<<<dim3(T), blk256, 0, stream>>>(ye, slot1, slot2, g1o, g2o, t2, D);
  add_ln_kernel<<<dim3(T), blk256, 0, stream>>>(x2, t2, ln3_g, ln3_b, (float*)d_out, D);
}
